// GCN_65747359367660
// MI455X (gfx1250) — compile-verified, matrix operand order checked by reading, not run on hardware
//
#include <hip/hip_runtime.h>

// ---------------------------------------------------------------------------
// Fused ST-GCN unit for MI455X (gfx1250, wave32, WMMA):
//   out[n] = sum_k (A[k]^T @ xf[n]) @ W[k]^T + S @ Bpad  (bias as a WMMA too)
// bf16 WMMA (f32 accumulate), 16x16x32 tiles; one workgroup (8 wave32) per n.
// Single merged LDS allocation -> manual layout -> DS-immediate addressing.
// ---------------------------------------------------------------------------

typedef __attribute__((ext_vector_type(16))) __bf16        v16bf;
typedef __attribute__((ext_vector_type(2)))  __bf16        v2bf;
typedef __attribute__((ext_vector_type(8)))  float         v8f;
typedef __attribute__((ext_vector_type(4)))  unsigned int  v4u;

#define Nn  1024
#define Cc  32
#define Tt  8
#define Vv  64
#define Kk  3
#define CT  256

#define AT_S 72    // At[k][w][v]: 64 rows/branch, stride 72 bf16 (144 B)
#define Z_S  264   // Z[w][c]    : 64 rows, stride 264 bf16 (528 B)
#define XT_S 72    // xfT[c][v]  : 256 rows, stride 72 bf16
#define SP_S 40    // spad[w][k] : 64 rows, stride 40 bf16 (K padded 3->32)
#define BP_S 40    // bpadT[d][k]: 256 rows, stride 40 bf16

#define AT_TOT (Kk * Vv * AT_S)   // 13824 elems @ elem-offset 0      (byte 0)
#define Z_TOT  (Vv * Z_S)         // 16896 elems @ 13824              (byte 27648)
#define XT_TOT (CT * XT_S)        // 18432 elems @ 30720              (byte 61440)
#define SP_TOT (Vv * SP_S)        //  2560 elems @ 49152              (byte 98304)
#define BP_TOT (CT * BP_S)        // 10240 elems @ 51712              (byte 103424)
#define LDS_TOT (AT_TOT + Z_TOT + XT_TOT + SP_TOT + BP_TOT)   // 61952 (121.0 KB)

// native fptrunc (RNE) -> v_cvt_pk_bf16_f32 on gfx1250
__device__ __forceinline__ unsigned int pk2bf(float lo, float hi) {
    v2bf t = { (__bf16)lo, (__bf16)hi };
    return __builtin_bit_cast(unsigned int, t);
}

union Frag {
    v16bf          v;
    v4u            q[2];
    unsigned int   d[8];
    unsigned short s[16];
};

__device__ __forceinline__ void ld_frag(Frag& f, const unsigned short* p) {
    f.q[0] = *(const v4u*)(p);        // K elements  8g..8g+7     (16 B)
    f.q[1] = *(const v4u*)(p + 16);   // K elements 16+8g..23+8g  (16 B)
}

__device__ __forceinline__ v8f wmma_bf16(const Frag& a, const Frag& b, v8f c) {
    return __builtin_amdgcn_wmma_f32_16x16x32_bf16(
        false, a.v, false, b.v, (short)0, c, false, false);
}

// one-shot: W fp32 -> bf16 into workspace (amortized over 1024 blocks of work)
__global__ __launch_bounds__(256)
void conv_w_bf16(const float* __restrict__ W, unsigned short* __restrict__ wbf) {
    int i = blockIdx.x * 256 + threadIdx.x;       // one fp32 pair per thread
    const float2 p = ((const float2*)W)[i];
    ((unsigned int*)wbf)[i] = pk2bf(p.x, p.y);
}

template <bool WBF16>
__global__ __launch_bounds__(256)
void gcn_fused_wmma(const float* __restrict__ x,            // (N,C,T,V)
                    const float* __restrict__ W,            // (K,CT,CT) fp32
                    const float* __restrict__ b,            // (K,CT)
                    const float* __restrict__ A,            // (K,V,V)
                    const unsigned short* __restrict__ wbf, // (K,CT,CT) bf16
                    float* __restrict__ out)                // (N,C,V,T)
{
    // single allocation: backend cannot reorder pieces, layout is ours
    __shared__ __align__(16) unsigned short lds[LDS_TOT];
    unsigned short* const atL   = lds;                       // hottest @ 0
    unsigned short* const zL    = lds + AT_TOT;
    unsigned short* const xfT   = lds + AT_TOT + Z_TOT;
    unsigned short* const spadL = lds + AT_TOT + Z_TOT + XT_TOT;
    unsigned short* const bpT   = lds + AT_TOT + Z_TOT + XT_TOT + SP_TOT;

    const int n    = blockIdx.x;
    const int tid  = threadIdx.x;
    const int wave = tid >> 5;
    const int lane = tid & 31;
    const int g    = lane >> 4;
    const int ln   = lane & 15;

    // ------------- stage 1: stage operands into LDS (transpose + cvt) -------
    const float* xn = x + (long)n * (CT * Vv);
    #pragma unroll 4
    for (int i = tid; i < (CT * Vv) / 2; i += 256) { // x[n] flat = c*64 + v
        int c = i >> 5, v = (i & 31) * 2;
        float2 p = *(const float2*)&xn[c * 64 + v];  // coalesced 8B reads
        *(unsigned int*)&xfT[c * XT_S + v] = pk2bf(p.x, p.y);
    }
    #pragma unroll 4
    for (int i = tid; i < (Kk * Vv * Vv) / 2; i += 256) {
        int k = i >> 11, r = i & 2047;
        int w = r & 63, v = (r >> 6) * 2;            // A flat = k*4096 + v*64 + w
        float a0 = A[k * 4096 + v * 64 + w];
        float a1 = A[k * 4096 + (v + 1) * 64 + w];
        *(unsigned int*)&atL[(k * Vv + w) * AT_S + v] = pk2bf(a0, a1);
    }
    // b^T with K padded to 32 (zero), u32 granularity => no write races
    #pragma unroll 1
    for (int i = tid; i < CT * (BP_S / 2); i += 256) {
        int d = i / (BP_S / 2), kq = i % (BP_S / 2);
        unsigned int val = 0u;
        if (kq == 0)      val = pk2bf(b[d], b[CT + d]);
        else if (kq == 1) val = pk2bf(b[2 * CT + d], 0.f);
        ((unsigned int*)bpT)[d * (BP_S / 2) + kq] = val;
    }
    // spad[w][k] = sum_v A[k][v][w], K padded to 32 (zero)
    #pragma unroll 1
    for (int i = tid; i < Vv * (SP_S / 2); i += 256) {
        int w = i / (SP_S / 2), kq = i % (SP_S / 2);
        unsigned int val = 0u;
        if (kq <= 1) {
            float s0 = 0.f, s1 = 0.f;
            if (kq == 0) {
                #pragma unroll 4
                for (int v = 0; v < Vv; ++v) { s0 += A[v * 64 + w]; s1 += A[4096 + v * 64 + w]; }
            } else {
                #pragma unroll 4
                for (int v = 0; v < Vv; ++v) { s0 += A[8192 + v * 64 + w]; }
            }
            val = pk2bf(s0, s1);
        }
        ((unsigned int*)spadL)[w * (SP_S / 2) + kq] = val;
    }
    // warm the (L2-resident) weights
    #pragma unroll 4
    for (int i = tid; i < (Kk * CT * CT) / 64; i += 256) {
        if (WBF16) __builtin_prefetch(wbf + i * 64, 0, 0);
        else       __builtin_prefetch(W + i * 64, 0, 0);
    }

    __syncthreads();

    const int cb0 = (wave * 2 + 0) * 16 + ln;   // lane column (stage2: c, stage3: d)
    const int cb1 = (wave * 2 + 1) * 16 + ln;

    // per-lane base pointers: everything else folds into DS immediate offsets
    const unsigned short* atBase = atL + ln * AT_S + g * 8;   // + k*Vv*AT_S + wt*16*AT_S + vs*32
    const unsigned short* zBaseA = zL + ln * Z_S + g * 8;     // + wt*16*Z_S + cs*32
    const unsigned short* xB0    = xfT + cb0 * XT_S + g * 8;  // + vs*32
    const unsigned short* xB1    = xfT + cb1 * XT_S + g * 8;
    unsigned short* zSt0 = zL + (8 * g) * Z_S + cb0;          // + (wt*16+r)*Z_S
    unsigned short* zSt1 = zL + (8 * g) * Z_S + cb1;

    v8f oacc[4][2] = {};

    #pragma unroll 1
    for (int k = 0; k < Kk; ++k) {
        // --------- stage 2: Z_k = A[k]^T @ xf  (64x64 @ 64x256) -------------
        v8f zacc[4][2] = {};
        const unsigned short* atk = atBase + k * (Vv * AT_S);
        #pragma unroll
        for (int vs = 0; vs < 2; ++vs) {             // K = 64 -> 2 steps
            Frag aF[4], bF[2];
            #pragma unroll
            for (int wt = 0; wt < 4; ++wt)
                ld_frag(aF[wt], atk + wt * 16 * AT_S + vs * 32);
            ld_frag(bF[0], xB0 + vs * 32);
            ld_frag(bF[1], xB1 + vs * 32);
            #pragma unroll
            for (int wt = 0; wt < 4; ++wt) {
                zacc[wt][0] = wmma_bf16(aF[wt], bF[0], zacc[wt][0]);
                zacc[wt][1] = wmma_bf16(aF[wt], bF[1], zacc[wt][1]);
            }
        }

        __syncthreads();   // previous iteration's zL readers are done
        #pragma unroll
        for (int wt = 0; wt < 4; ++wt)
            #pragma unroll
            for (int r = 0; r < 8; r += 2) {         // packed bf16 converts
                unsigned int p0 = pk2bf(zacc[wt][0][r], zacc[wt][0][r + 1]);
                unsigned int p1 = pk2bf(zacc[wt][1][r], zacc[wt][1][r + 1]);
                zSt0[(wt * 16 + r) * Z_S]       = (unsigned short)p0;
                zSt0[(wt * 16 + r + 1) * Z_S]   = (unsigned short)(p0 >> 16);
                zSt1[(wt * 16 + r) * Z_S]       = (unsigned short)p1;
                zSt1[(wt * 16 + r + 1) * Z_S]   = (unsigned short)(p1 >> 16);
            }
        __syncthreads();   // Z_k visible to all waves

        // --------- stage 3: out += Z_k @ W[k]^T  (64x256 @ 256x256) ---------
        const unsigned short* wb0 = WBF16 ? (wbf + (long)k * (CT * CT) + (long)cb0 * CT + g * 8) : nullptr;
        const unsigned short* wb1 = WBF16 ? (wbf + (long)k * (CT * CT) + (long)cb1 * CT + g * 8) : nullptr;
        const float* wk = W + (long)k * (CT * CT);

        #pragma unroll
        for (int cs = 0; cs < 8; ++cs) {             // K = 256 -> 8 steps
            Frag aF[4], bF[2];
            #pragma unroll
            for (int wt = 0; wt < 4; ++wt)
                ld_frag(aF[wt], zBaseA + wt * 16 * Z_S + cs * 32);
            if constexpr (WBF16) {
                ld_frag(bF[0], wb0 + cs * 32);
                ld_frag(bF[1], wb1 + cs * 32);
            } else {
                #pragma unroll
                for (int dc = 0; dc < 2; ++dc) {
                    int d = dc ? cb1 : cb0;
                    const float* wrow = wk + (long)d * CT;
                    #pragma unroll
                    for (int j = 0; j < 8; ++j) {
                        int c = cs * 32 + ((j & 4) << 2) + ((j & 3) << 1) + (g << 3);
                        bF[dc].d[j] = pk2bf(wrow[c], wrow[c + 1]);
                    }
                }
            }
            #pragma unroll
            for (int wt = 0; wt < 4; ++wt) {
                oacc[wt][0] = wmma_bf16(aF[wt], bF[0], oacc[wt][0]);
                oacc[wt][1] = wmma_bf16(aF[wt], bF[1], oacc[wt][1]);
            }
        }
    }

    // ------------- bias as one zero-padded WMMA pass: oacc += S @ Bpad ------
    {
        Frag sF[4], bF[2];
        #pragma unroll
        for (int wt = 0; wt < 4; ++wt)
            ld_frag(sF[wt], spadL + (wt * 16 + ln) * SP_S + 8 * g);
        ld_frag(bF[0], bpT + cb0 * BP_S + 8 * g);
        ld_frag(bF[1], bpT + cb1 * BP_S + 8 * g);
        #pragma unroll
        for (int wt = 0; wt < 4; ++wt) {
            oacc[wt][0] = wmma_bf16(sF[wt], bF[0], oacc[wt][0]);
            oacc[wt][1] = wmma_bf16(sF[wt], bF[1], oacc[wt][1]);
        }
    }

    // ------------- epilogue: pure stores, (N,V,CT) -> (N,C,V,T) permute -----
    #pragma unroll
    for (int dc = 0; dc < 2; ++dc) {
        int d  = dc ? cb1 : cb0;
        int cO = d >> 3, t = d & 7;                  // d = c*T + t, T = 8
        float* op = out + (((long)n * Cc + cO) * Vv + 8 * g) * Tt + t;
        #pragma unroll
        for (int wt = 0; wt < 4; ++wt)
            #pragma unroll
            for (int r = 0; r < 8; ++r)
                op[(wt * 16 + r) * Tt] = oacc[wt][dc][r];
    }
}

extern "C" void kernel_launch(void* const* d_in, const int* in_sizes, int n_in,
                              void* d_out, int out_size, void* d_ws, size_t ws_size,
                              hipStream_t stream) {
    (void)in_sizes; (void)n_in; (void)out_size;
    const float* x = (const float*)d_in[0];
    const float* W = (const float*)d_in[1];
    const float* b = (const float*)d_in[2];
    const float* A = (const float*)d_in[3];
    float* out = (float*)d_out;

    const size_t wbf_bytes = (size_t)Kk * CT * CT * sizeof(unsigned short);
    if (ws_size >= wbf_bytes) {
        unsigned short* wbf = (unsigned short*)d_ws;
        conv_w_bf16<<<dim3((Kk * CT * CT / 2) / 256), dim3(256), 0, stream>>>(W, wbf);
        gcn_fused_wmma<true><<<dim3(Nn), dim3(256), 0, stream>>>(x, W, b, A, wbf, out);
    } else {
        gcn_fused_wmma<false><<<dim3(Nn), dim3(256), 0, stream>>>(x, W, b, A, nullptr, out);
    }
}
